// Pillar_Encoder_87376814670471
// MI455X (gfx1250) — matrix operand order repaired and checked by reading, not compiled, not run on hardware
//
#include <hip/hip_runtime.h>
#include <hip/hip_bf16.h>

typedef __attribute__((ext_vector_type(16))) _Float16 v16h;
typedef __attribute__((ext_vector_type(8)))  _Float16 v8h;
typedef __attribute__((ext_vector_type(8)))  float    v8f;
typedef __attribute__((ext_vector_type(4)))  int      v4i;

#define BATCH 8
#define NPTS  20000
#define NPIL  100
#define TILE  64
#define TPB   313   // ceil(20000/64)

// CDNA5 async LDS<->global path (gfx1250), guarded so compile never regresses.
#if defined(__has_builtin)
#if __has_builtin(__builtin_amdgcn_global_load_async_to_lds_b128)
#define HAVE_ASYNC_LD 1
#endif
#if __has_builtin(__builtin_amdgcn_global_store_async_from_lds_b128)
#define HAVE_ASYNC_ST 1
#endif
#endif

// Builtin prototype (from hipcc diagnostic): (v4i AS1*, v4i AS3*, Imm int, Imm int)
typedef __attribute__((address_space(1))) v4i as1_v4i;
typedef __attribute__((address_space(3))) v4i as3_v4i;

__device__ __forceinline__ void async_wait0() {
#if defined(__has_builtin)
#if __has_builtin(__builtin_amdgcn_s_wait_asynccnt)
    __builtin_amdgcn_s_wait_asynccnt(0);
#else
    asm volatile("s_wait_asynccnt 0x0" ::: "memory");
#endif
#else
    asm volatile("s_wait_asynccnt 0x0" ::: "memory");
#endif
}

// ---------------------------------------------------------------------------
// WMMA fragment loaders (CDNA5 16x16x32 f16 layouts, wave32)
// A (16x32, MxK): m=lane&15, hi=lane>>4; halves j=0..7 -> K=kbase+hi*8+j,
//                 j=8..15 -> K=kbase+16+hi*8+(j-8)  => two contiguous b128 loads
// B (32x16, KxN): n=lane&15, hi=lane>>4; half j -> K=(j>>3)*16+hi*8+(j&7) (repacked so each
//                 lane's 16 halves are 32 contiguous bytes in memory)
// C/D (16x16 f32): VGPR r -> M = r + 8*hi, N = lane&15
// ---------------------------------------------------------------------------
__device__ __forceinline__ v16h ldsAfrag(const _Float16* src, int strideH,
                                         int rowBase, int lane, int kstep) {
    int m = lane & 15, hi = lane >> 4;
    const _Float16* rp = src + (size_t)(rowBase + m) * strideH + kstep * 32 + hi * 8;
    v8h lo = *(const v8h*)rp;
    v8h hh = *(const v8h*)(rp + 16);
    v16h a;
#pragma unroll
    for (int j = 0; j < 8; ++j) { a[j] = lo[j]; a[j + 8] = hh[j]; }
    return a;
}

__device__ __forceinline__ v16h gblBfrag(const _Float16* frags, int kstep,
                                         int ct, int nCT, int lane) {
    const _Float16* p = frags + ((size_t)(((kstep * nCT) + ct) * 32 + lane) << 4);
    v8h lo = *(const v8h*)p;
    v8h hh = *(const v8h*)(p + 8);
    v16h b;
#pragma unroll
    for (int j = 0; j < 8; ++j) { b[j] = lo[j]; b[j + 8] = hh[j]; }
    return b;
}

// One fused MLP layer over a 64-row LDS tile: dst = relu(src @ W), f16 in/out.
// 4 row-tiles x nCT col-tiles distributed over 8 waves.
__device__ __forceinline__ void mlp_layer(const _Float16* src, int srcStride,
                                          _Float16* dst, int dstStride,
                                          const _Float16* wfr, int nCT, int nK,
                                          int wave, int lane) {
    const int nTiles = 4 * nCT;
    for (int T = wave; T < nTiles; T += 8) {
        int rt = T / nCT, ct = T - rt * nCT;
        v8f acc = {};
        for (int k = 0; k < nK; ++k) {
            v16h a = ldsAfrag(src, srcStride, rt * 16, lane, k);
            v16h b = gblBfrag(wfr, k, ct, nCT, lane);
            acc = __builtin_amdgcn_wmma_f32_16x16x32_f16(
                false, a, false, b, (short)0, acc, false, false);
        }
        int n = lane & 15, hi = lane >> 4;
#pragma unroll
        for (int r = 0; r < 8; ++r) {
            float v = fmaxf(acc[r], 0.0f);
            dst[(size_t)(rt * 16 + r + 8 * hi) * dstStride + ct * 16 + n] = (_Float16)v;
        }
    }
}

// ---------------------------------------------------------------------------
__global__ __launch_bounds__(256) void k_zero(float* p, int n) {
    int i = blockIdx.x * blockDim.x + threadIdx.x;
    if (i < n) p[i] = 0.0f;
}

// Repack row-major f32 W[Kin x Nout] into f16 B-fragment layout
__global__ __launch_bounds__(256) void k_repack(const float* __restrict__ src,
                                                _Float16* __restrict__ dst,
                                                int Kin, int Nout, int nCT, int total) {
    int e = blockIdx.x * blockDim.x + threadIdx.x;
    if (e >= total) return;
    int j = e & 15;
    int lane = (e >> 4) & 31;
    int ct = (e >> 9) % nCT;
    int k  = (e >> 9) / nCT;
    int K = k * 32 + (j >> 3) * 16 + (lane >> 4) * 8 + (j & 7);
    int n = ct * 16 + (lane & 15);
    dst[e] = (K < Kin) ? (_Float16)src[(size_t)K * Nout + n] : (_Float16)0.0f;
}

// Per-point pillar index + centroid accumulation (atomics)
__global__ __launch_bounds__(256) void k_points(const float* __restrict__ x,
                                                int* __restrict__ segg,
                                                float* __restrict__ counts,
                                                float* __restrict__ sums) {
    int i = blockIdx.x * blockDim.x + threadIdx.x;
    if (i >= BATCH * NPTS) return;
    const float* xp = x + (size_t)i * 3;
    float x0 = xp[0], x1 = xp[1], x2 = xp[2];
    // pts = (x1, x0, x2); seg from (x0, x2)
    float yy = fminf(fmaxf(x0 + 1.0f, 0.0f), 1.99f);
    float zz = fminf(fmaxf(x2 + 1.0f, 0.0f), 1.99f);
    int seg = ((int)floorf(yy * 5.0f)) * 10 + (int)floorf(zz * 5.0f);
    segg[i] = seg;
    int b = i / NPTS;
    int base = b * NPIL + seg;
    atomicAdd(&counts[base], 1.0f);
    atomicAdd(&sums[base * 3 + 0], x1);
    atomicAdd(&sums[base * 3 + 1], x0);
    atomicAdd(&sums[base * 3 + 2], x2);
}

// Fused MLP layers 1-3 per 64-point tile; h3 -> global f16 + pillar-max (f32 bits)
__global__ __launch_bounds__(256) void k_mlp123(
    const float* __restrict__ x, const int* __restrict__ segg,
    const float* __restrict__ counts, const float* __restrict__ sums,
    const _Float16* __restrict__ W1r, const _Float16* __restrict__ W2r,
    const _Float16* __restrict__ W3r,
    _Float16* __restrict__ h3g, unsigned* __restrict__ pilU) {
    __shared__ _Float16 sAug[TILE * 32];
    __shared__ _Float16 sH1[TILE * 32];
    __shared__ _Float16 sH2[TILE * 64];
    __shared__ _Float16 sH3[TILE * 128];
    __shared__ int sSeg[TILE];

    const int b = blockIdx.x / TPB;
    const int t = blockIdx.x % TPB;
    const int n0 = t * TILE;
    const int valid = min(TILE, NPTS - n0);
    const int tid = threadIdx.x;

    if (tid < TILE) {
        int r = tid;
        _Float16* ar = sAug + r * 32;
        if (r < valid) {
            size_t pi = (size_t)(b * NPTS + n0 + r);
            const float* xp = x + pi * 3;
            float p0 = xp[1], p1 = xp[0], p2 = xp[2];   // swapped (Z,Y,X)
            int seg = segg[pi];
            sSeg[r] = seg;
            int cb = b * NPIL + seg;
            float cnt = fmaxf(counts[cb], 1.0f);
            float c0 = sums[cb * 3 + 0] / cnt;
            float c1 = sums[cb * 3 + 1] / cnt;
            float c2 = sums[cb * 3 + 2] / cnt;
            ar[0] = (_Float16)p0; ar[1] = (_Float16)p1; ar[2] = (_Float16)p2;
            ar[3] = (_Float16)(p0 - c0); ar[4] = (_Float16)(p1 - c1); ar[5] = (_Float16)(p2 - c2);
#pragma unroll
            for (int j = 6; j < 32; ++j) ar[j] = (_Float16)0.0f;
        } else {
            sSeg[r] = 0;
#pragma unroll
            for (int j = 0; j < 32; ++j) ar[j] = (_Float16)0.0f;
        }
    }
    __syncthreads();
    int wave = tid >> 5, lane = tid & 31;
    mlp_layer(sAug, 32, sH1, 32, W1r, 2, 1, wave, lane);   // 6(pad32)->32
    __syncthreads();
    mlp_layer(sH1, 32, sH2, 64, W2r, 4, 1, wave, lane);    // 32->64
    __syncthreads();
    mlp_layer(sH2, 64, sH3, 128, W3r, 8, 2, wave, lane);   // 64->128
    __syncthreads();
#if HAVE_ASYNC_ST
    // h3 tile write-out: LDS -> global, no VGPR round-trip (ASYNCcnt path).
    if (tid < TILE && tid < valid) {
        unsigned  lbase = (unsigned)(uintptr_t)(sH3 + tid * 128);
        uintptr_t gbase = (uintptr_t)(h3g + ((size_t)(b * NPTS + n0 + tid)) * 128);
#pragma unroll
        for (int q = 0; q < 16; ++q)
            __builtin_amdgcn_global_store_async_from_lds_b128(
                (as1_v4i*)(gbase + q * 16), (as3_v4i*)(uintptr_t)(lbase + q * 16), 0, 0);
    }
    for (int e = tid; e < TILE * 128; e += 256) {
        int r = e >> 7, c = e & 127;
        if (r < valid) {
            atomicMax(&pilU[((size_t)(b * NPIL + sSeg[r])) * 128 + c],
                      __float_as_uint((float)sH3[e]));
        }
    }
    async_wait0();   // LDS must not be reused before stores drain (endpgm also waits)
#else
    for (int e = tid; e < TILE * 128; e += 256) {
        int r = e >> 7, c = e & 127;
        if (r < valid) {
            _Float16 hv = sH3[e];
            h3g[((size_t)(b * NPTS + n0 + r)) * 128 + c] = hv;
            atomicMax(&pilU[((size_t)(b * NPIL + sSeg[r])) * 128 + c],
                      __float_as_uint((float)hv));
        }
    }
#endif
}

// Layer 4: [h | pil[seg]] (256) @ W4 (256x768), relu, segment-max into enc
__global__ __launch_bounds__(256) void k_layer4(
    const _Float16* __restrict__ h3g, const int* __restrict__ segg,
    const float* __restrict__ pilF, const _Float16* __restrict__ W4r,
    unsigned* __restrict__ encU) {
    __shared__ _Float16 sA[TILE * 264];   // 256 halves + 8 pad, 16B-aligned rows
    __shared__ int sSeg[TILE];

    const int b = blockIdx.x / TPB;
    const int t = blockIdx.x % TPB;
    const int n0 = t * TILE;
    const int valid = min(TILE, NPTS - n0);
    const int tid = threadIdx.x;

    if (tid < TILE) {
        int r = tid;
        _Float16* row = sA + r * 264;
        if (r < valid) {
            size_t pi = (size_t)(b * NPTS + n0 + r);
            int seg = segg[pi];
            sSeg[r] = seg;
#if HAVE_ASYNC_LD
            // h row: global -> LDS directly (ASYNCcnt path, no VGPR round-trip)
            uintptr_t gbase = (uintptr_t)(h3g + pi * 128);
            unsigned  lbase = (unsigned)(uintptr_t)row;
#pragma unroll
            for (int q = 0; q < 16; ++q)
                __builtin_amdgcn_global_load_async_to_lds_b128(
                    (as1_v4i*)(gbase + q * 16), (as3_v4i*)(uintptr_t)(lbase + q * 16), 0, 0);
#else
            const uint4* hp = (const uint4*)(h3g + pi * 128);
            uint4* dp = (uint4*)row;
#pragma unroll
            for (int q = 0; q < 16; ++q) dp[q] = hp[q];
#endif
            const float* pp = pilF + ((size_t)(b * NPIL + seg)) * 128;
#pragma unroll 4
            for (int c = 0; c < 128; ++c) row[128 + c] = (_Float16)pp[c];
        } else {
            sSeg[r] = 0;
            for (int c = 0; c < 256; ++c) row[c] = (_Float16)0.0f;
        }
    }
#if HAVE_ASYNC_LD
    async_wait0();
#endif
    __syncthreads();
    int wave = tid >> 5, lane = tid & 31;
    int n = lane & 15, hi = lane >> 4;
    for (int rt = 0; rt < 4; ++rt) {
        // Hoist A fragments: reused across the 6 col-tiles this wave owns.
        v16h a[8];
#pragma unroll
        for (int k = 0; k < 8; ++k) a[k] = ldsAfrag(sA, 264, rt * 16, lane, k);
        for (int ct = wave; ct < 48; ct += 8) {
            v8f acc = {};
#pragma unroll
            for (int k = 0; k < 8; ++k) {           // K = 256 = 8 x 32
                v16h bm = gblBfrag(W4r, k, ct, 48, lane);
                acc = __builtin_amdgcn_wmma_f32_16x16x32_f16(
                    false, a[k], false, bm, (short)0, acc, false, false);
            }
#pragma unroll
            for (int r8 = 0; r8 < 8; ++r8) {
                int tr = rt * 16 + r8 + 8 * hi;
                if (tr < valid) {
                    float v = fmaxf(acc[r8], 0.0f); // >=0 => uint-bit max == float max
                    atomicMax(&encU[((size_t)(b * NPIL + sSeg[tr])) * 768 + ct * 16 + n],
                              __float_as_uint(v));
                }
            }
        }
    }
}

// BatchNorm1d(100) training-mode stats over dims (B, 768), one block per pillar
__global__ __launch_bounds__(256) void k_bn(const float* __restrict__ enc,
                                            const float* __restrict__ gamma,
                                            const float* __restrict__ beta,
                                            float* __restrict__ out) {
    const int pid = blockIdx.x, tid = threadIdx.x;
    __shared__ float s1[256], s2[256];
    __shared__ float sMean, sInv;
    float sum = 0.0f, sq = 0.0f;
    for (int e = tid; e < BATCH * 768; e += 256) {
        int bb = e / 768, c = e % 768;
        float v = enc[((size_t)(bb * NPIL + pid)) * 768 + c];
        sum += v; sq += v * v;
    }
    s1[tid] = sum; s2[tid] = sq;
    __syncthreads();
    for (int off = 128; off > 0; off >>= 1) {
        if (tid < off) { s1[tid] += s1[tid + off]; s2[tid] += s2[tid + off]; }
        __syncthreads();
    }
    if (tid == 0) {
        float m = s1[0] / (float)(BATCH * 768);
        float var = s2[0] / (float)(BATCH * 768) - m * m;
        sMean = m; sInv = rsqrtf(var + 1e-5f);
    }
    __syncthreads();
    float g = gamma[pid], bt = beta[pid], m = sMean, inv = sInv;
    for (int e = tid; e < BATCH * 768; e += 256) {
        int bb = e / 768, c = e % 768;
        size_t ix = ((size_t)(bb * NPIL + pid)) * 768 + c;
        out[ix] = (enc[ix] - m) * inv * g + bt;
    }
}

// ---------------------------------------------------------------------------
extern "C" void kernel_launch(void* const* d_in, const int* in_sizes, int n_in,
                              void* d_out, int out_size, void* d_ws, size_t ws_size,
                              hipStream_t stream) {
    const float* x     = (const float*)d_in[0];
    const float* W1    = (const float*)d_in[1];
    const float* W2    = (const float*)d_in[2];
    const float* W3    = (const float*)d_in[3];
    const float* W4    = (const float*)d_in[4];
    const float* gamma = (const float*)d_in[5];
    const float* beta  = (const float*)d_in[6];
    float* out = (float*)d_out;

    char* ws = (char*)d_ws;
    // [counts 800f | sums 2400f | pil 102400u | enc 614400u] contiguous (zeroed)
    float*    counts = (float*)(ws + 0);
    float*    sums   = (float*)(ws + 3200);
    unsigned* pilU   = (unsigned*)(ws + 12800);
    float*    pilF   = (float*)(ws + 12800);
    unsigned* encU   = (unsigned*)(ws + 422400);
    float*    encF   = (float*)(ws + 422400);
    int*      segg   = (int*)(ws + 2880000);        // 160000 ints
    _Float16* h3g    = (_Float16*)(ws + 3520000);   // 160000*128 halves
    _Float16* W1r    = (_Float16*)(ws + 44480000);  // 1024 halves
    _Float16* W2r    = (_Float16*)(ws + 44482048);  // 2048
    _Float16* W3r    = (_Float16*)(ws + 44486144);  // 8192
    _Float16* W4r    = (_Float16*)(ws + 44502528);  // 196608

    k_zero<<<(720000 + 255) / 256, 256, 0, stream>>>((float*)ws, 720000);
    k_repack<<<(1024   + 255) / 256, 256, 0, stream>>>(W1, W1r, 6,   32,  2,  1024);
    k_repack<<<(2048   + 255) / 256, 256, 0, stream>>>(W2, W2r, 32,  64,  4,  2048);
    k_repack<<<(8192   + 255) / 256, 256, 0, stream>>>(W3, W3r, 64,  128, 8,  8192);
    k_repack<<<(196608 + 255) / 256, 256, 0, stream>>>(W4, W4r, 256, 768, 48, 196608);
    k_points<<<(BATCH * NPTS + 255) / 256, 256, 0, stream>>>(x, segg, counts, sums);
    k_mlp123<<<BATCH * TPB, 256, 0, stream>>>(x, segg, counts, sums, W1r, W2r, W3r, h3g, pilU);
    k_layer4<<<BATCH * TPB, 256, 0, stream>>>(h3g, segg, pilF, W4r, encU);
    k_bn<<<NPIL, 256, 0, stream>>>(encF, gamma, beta, out);
}
